// AdaptiveConv_40699110097793
// MI455X (gfx1250) — compile-verified, matrix operand order checked by reading, not run on hardware
//
#include <hip/hip_runtime.h>
#include <hip/hip_bf16.h>

typedef __attribute__((ext_vector_type(16))) __bf16 v16bf;
typedef __attribute__((ext_vector_type(8)))  __bf16 v8bf;
typedef __attribute__((ext_vector_type(8)))  float  v8f;

#define NB     4
#define C_IN   256
#define HH     64
#define WW     64
#define OC     256
#define HW     4096
#define KTOT   2304     // C*9
#define NCHUNK 72       // KTOT / 32
#define BPITCH 40       // bf16 elems per B row (80B: 16B aligned, conflict-free stride)

// ---------------------------------------------------------------------------
// Pre-pass: fp32 weight [O][C][3][3] -> bf16 A-fragments in WMMA lane order.
// Flat layout: [kc(72)][mt(16)][lane(32)][i(16)] so the main kernel loads one
// v16bf (32B) per lane with a single coalesced global access.
// Fragment map (16-bit A 16x32): lane<16 holds K{0..7,16..23}, lane>=16 holds
// K{8..15,24..31}; element i<8 -> first K group, i>=8 -> second.
// ---------------------------------------------------------------------------
__global__ __launch_bounds__(256) void prep_weight_bf16(
    const float* __restrict__ w, __bf16* __restrict__ wbf) {
  int t    = blockIdx.x * 256 + threadIdx.x;      // < 72*16*32*16 = 589824
  int kc   = t >> 13;                             // /8192
  int r    = t & 8191;
  int mt   = r >> 9;                              // /512
  int r2   = r & 511;
  int lane = r2 >> 4;
  int i    = r2 & 15;
  int k    = (i < 8 ? i : i + 8) + ((lane >= 16) ? 8 : 0);   // k-local 0..31
  int m    = mt * 16 + (lane & 15);
  int kpos = kc >> 3;                             // tap 0..8
  int c    = ((kc & 7) << 5) + k;                 // channel
  wbf[t] = (__bf16)w[(size_t)m * KTOT + c * 9 + kpos];
}

// ---------------------------------------------------------------------------
// Main kernel: one WG per (n, output row h). Produces out[n, 0..255, h, 0..63].
// Software-pipelined: gathers for chunk kc+1 are in flight while the WMMAs for
// chunk kc execute; the loadcnt wait lands after the matrix ops.
// ---------------------------------------------------------------------------
__global__ __launch_bounds__(256) void dconv_wmma_kernel(
    const float* __restrict__ x, const float* __restrict__ off,
    const __bf16* __restrict__ wbf, float* __restrict__ out) {
  __shared__ float  smeta[576][8];                 // 9 taps x 64 cols: w00..w11, o00..o11
  __shared__ __bf16 Bs[2][64 * BPITCH];            // double-buffered val tile [pos][c]

  const int tid  = threadIdx.x;
  const int h    = blockIdx.x;
  const int n    = blockIdx.y;
  const int lane = tid & 31;
  const int wv   = tid >> 5;                       // wave id 0..7
  const int pg   = wv;                             // producer position group

  // ---- bilinear metadata: shared across all 256 channels ----
  for (int e = tid; e < 576; e += 256) {
    int   kpos = e >> 6;
    int   wcol = e & 63;
    const float* op = off + ((size_t)(n * HW + h * WW + wcol)) * 18 + kpos * 2;
    float dy = op[0];
    float dx = op[1];
    float py = (float)(h + (kpos / 3) - 1) + dy;
    float px = (float)(wcol + (kpos % 3) - 1) + dx;
    float y0f = floorf(py), x0f = floorf(px);
    int   y0 = (int)y0f, x0 = (int)x0f, y1 = y0 + 1, x1 = x0 + 1;
    float fy = py - y0f, fx = px - x0f;
    bool  vy0 = (y0 >= 0) && (y0 < HH), vy1 = (y1 >= 0) && (y1 < HH);
    bool  vx0 = (x0 >= 0) && (x0 < WW), vx1 = (x1 >= 0) && (x1 < WW);
    int   cy0 = y0 < 0 ? 0 : (y0 > HH - 1 ? HH - 1 : y0);
    int   cy1 = y1 < 0 ? 0 : (y1 > HH - 1 ? HH - 1 : y1);
    int   cx0 = x0 < 0 ? 0 : (x0 > WW - 1 ? WW - 1 : x0);
    int   cx1 = x1 < 0 ? 0 : (x1 > WW - 1 ? WW - 1 : x1);
    smeta[e][0] = (vy0 && vx0) ? (1.f - fy) * (1.f - fx) : 0.f;
    smeta[e][1] = (vy0 && vx1) ? (1.f - fy) * fx         : 0.f;
    smeta[e][2] = (vy1 && vx0) ? fy * (1.f - fx)         : 0.f;
    smeta[e][3] = (vy1 && vx1) ? fy * fx                 : 0.f;
    smeta[e][4] = __int_as_float(cy0 * WW + cx0);
    smeta[e][5] = __int_as_float(cy0 * WW + cx1);
    smeta[e][6] = __int_as_float(cy1 * WW + cx0);
    smeta[e][7] = __int_as_float(cy1 * WW + cx1);
  }
  __syncthreads();

  // Phase 1 of producer: issue 32 scattered gather loads for chunk kc into regs.
  auto gather_issue = [&](int kc, float g[8][4]) {
    int kpos = kc >> 3;
    const float* xb = x + (((size_t)n * C_IN + ((kc & 7) << 5) + lane) << 12);
#pragma unroll
    for (int j = 0; j < 8; ++j) {
      const float* me = smeta[kpos * 64 + (pg * 8 + j)];
      float4 of = *(const float4*)(me + 4);
      g[j][0] = xb[__float_as_int(of.x)];
      g[j][1] = xb[__float_as_int(of.y)];
      g[j][2] = xb[__float_as_int(of.z)];
      g[j][3] = xb[__float_as_int(of.w)];
    }
  };
  // Phase 2 of producer: weighted combine + bf16 store into LDS B tile.
  auto combine_store = [&](int kc, int buf, const float g[8][4]) {
    int kpos = kc >> 3;
#pragma unroll
    for (int j = 0; j < 8; ++j) {
      int pos = pg * 8 + j;
      float4 wt = *(const float4*)(smeta[kpos * 64 + pos]);
      float v = g[j][0] * wt.x + g[j][1] * wt.y + g[j][2] * wt.z + g[j][3] * wt.w;
      Bs[buf][pos * BPITCH + lane] = (__bf16)v;
    }
  };

  v8f acc[2][4];
#pragma unroll
  for (int mi = 0; mi < 2; ++mi)
#pragma unroll
    for (int t = 0; t < 4; ++t)
      acc[mi][t] = (v8f){0.f, 0.f, 0.f, 0.f, 0.f, 0.f, 0.f, 0.f};

  const v16bf* af = (const v16bf*)wbf;
  // B-fragment base: col n = lane&15 -> row in tile; lane>=16 -> K{8..15,24..31}
  const int bbase = (lane & 15) * BPITCH + ((lane >> 4) << 3);

  float g[8][4];
  gather_issue(0, g);
  combine_store(0, 0, g);
  __syncthreads();

  for (int kc = 0; kc < NCHUNK; ++kc) {
    // A fragments for this chunk first (coalesced 32B/lane; ordered before the
    // gathers so waiting on them does not drain the gather loads).
    v16bf a0 = af[(size_t)(kc * 16 + wv * 2 + 0) * 32 + lane];
    v16bf a1 = af[(size_t)(kc * 16 + wv * 2 + 1) * 32 + lane];

    const bool more = (kc + 1) < NCHUNK;
    if (more) gather_issue(kc + 1, g);     // in flight during the WMMAs below

    const __bf16* bb = &Bs[kc & 1][0];
#pragma unroll
    for (int t = 0; t < 4; ++t) {
      const __bf16* bp = bb + t * (16 * BPITCH) + bbase;
      v8bf lo = *(const v8bf*)bp;          // K 0..7  (or 8..15)
      v8bf hi = *(const v8bf*)(bp + 16);   // K 16..23 (or 24..31)
      v16bf b;
#pragma unroll
      for (int q = 0; q < 8; ++q) { b[q] = lo[q]; b[q + 8] = hi[q]; }
      acc[0][t] = __builtin_amdgcn_wmma_f32_16x16x32_bf16(
          false, a0, false, b, (short)0, acc[0][t], false, false);
      acc[1][t] = __builtin_amdgcn_wmma_f32_16x16x32_bf16(
          false, a1, false, b, (short)0, acc[1][t], false, false);
    }

    if (more) combine_store(kc + 1, (kc + 1) & 1, g);  // loadcnt wait lands here
    __syncthreads();
  }

  // C/D layout: VGPR v -> row M = v + 8*(lane>=16); col N = lane&15.
  float* ob = out + (size_t)n * OC * HW + h * WW;
#pragma unroll
  for (int mi = 0; mi < 2; ++mi) {
    int orow = (wv * 2 + mi) * 16 + ((lane >> 4) << 3);
#pragma unroll
    for (int t = 0; t < 4; ++t) {
      int wcol = t * 16 + (lane & 15);
#pragma unroll
      for (int v = 0; v < 8; ++v)
        ob[(size_t)(orow + v) * HW + wcol] = acc[mi][t][v];
    }
  }
}

extern "C" void kernel_launch(void* const* d_in, const int* in_sizes, int n_in,
                              void* d_out, int out_size, void* d_ws, size_t ws_size,
                              hipStream_t stream) {
  (void)in_sizes; (void)n_in; (void)out_size; (void)ws_size;
  const float* x   = (const float*)d_in[0];
  const float* off = (const float*)d_in[1];
  const float* w   = (const float*)d_in[2];
  float*  out = (float*)d_out;
  __bf16* wbf = (__bf16*)d_ws;   // 72*16*32*16 bf16 = 1.18 MB

  prep_weight_bf16<<<2304, 256, 0, stream>>>(w, wbf);          // 589824 elems
  dconv_wmma_kernel<<<dim3(64, 4), 256, 0, stream>>>(x, off, wbf, out);
}
